// DenseAttentionOneHead_26345329393767
// MI455X (gfx1250) — matrix operand order repaired and checked by reading, not compiled, not run on hardware
//
#include <hip/hip_runtime.h>

// ---------------------------------------------------------------------------
// out = X @ W^T @ X^T @ X  ==  X @ (W^T @ (X^T @ X))   (associativity)
// Stage 1: G  = X^T X      (TN gemm, M=N=1024, K=4096, per batch)
// Stage 2: M2 = W^T G      (TN gemm, M=N=K=1024, per batch; W shared)
// Stage 3: out= X  M2      (NN gemm, M=4096, N=K=1024, per batch)
// fp32 emulated via bf16 hi/lo split: a*b ~= ah*bh + ah*bl + al*bh
// (3x v_wmma_f32_16x16x32_bf16 per tile-pair, fp32 accumulate)
// Software-pipelined: next K-step's global tile is staged in VGPRs while the
// current K-step's WMMAs execute, hiding HBM/L2 latency inside the wave.
// ---------------------------------------------------------------------------

typedef __attribute__((ext_vector_type(16))) __bf16 v16bf;
typedef __attribute__((ext_vector_type(8)))  float  v8f;

__device__ __forceinline__ unsigned short f2bf(float f) {
    unsigned u = __builtin_bit_cast(unsigned, f);
    u += 0x7fffu + ((u >> 16) & 1u);              // round-to-nearest-even
    return (unsigned short)(u >> 16);
}
__device__ __forceinline__ float bf2f(unsigned short h) {
    unsigned u = ((unsigned)h) << 16;
    return __builtin_bit_cast(float, u);
}

// LDS tile row stride in ushorts: 32 K-elems + 8 pad = 40 (80 B, 16B-aligned)
#define LDSTRIDE 40

// A fragment (16x32 bf16): lane half h, elems 0..7 -> K = 8h+e ; 8..15 -> K = 16+8h+(e-8)
__device__ __forceinline__ v16bf load_frag_a(const unsigned short* row, int hf) {
    uint4 x = *reinterpret_cast<const uint4*>(row + hf * 8);
    uint4 y = *reinterpret_cast<const uint4*>(row + hf * 8 + 16);
    union { uint4 q[2]; v16bf v; } u;
    u.q[0] = x; u.q[1] = y;
    return u.v;
}
// B fragment (32x16 bf16): lane half h, elems 0..7 -> K = 8h+e ; 8..15 -> K = 16+8h+(e-8)
__device__ __forceinline__ v16bf load_frag_b(const unsigned short* row, int hf) {
    uint4 x = *reinterpret_cast<const uint4*>(row + hf * 8);
    uint4 y = *reinterpret_cast<const uint4*>(row + hf * 8 + 16);
    union { uint4 q[2]; v16bf v; } u;
    u.q[0] = x; u.q[1] = y;
    return u.v;
}

// Stage next global K-slab into VGPRs (4 float4 for A, 4 for B per thread).
template <bool TRANS_A>
__device__ __forceinline__ void stage_tiles(
    const float* __restrict__ A, const float* __restrict__ B,
    int kt, int m0, int n0, int lda, int ldb, int tid,
    float4 (&stA)[4], float4 (&stB)[4]) {
#pragma unroll
    for (int i = 0; i < 4; ++i) {
        int f = tid + i * 256;
        if constexpr (TRANS_A) {
            int kr = f >> 5;              // 0..31
            int mf = (f & 31) << 2;       // 0..124
            stA[i] = *reinterpret_cast<const float4*>(
                A + (size_t)(kt + kr) * lda + m0 + mf);
        } else {
            int mr = f >> 3;              // 0..127
            int kf = (f & 7) << 2;        // 0..28
            stA[i] = *reinterpret_cast<const float4*>(
                A + (size_t)(m0 + mr) * lda + kt + kf);
        }
        int kr = f >> 5;
        int nf = (f & 31) << 2;
        stB[i] = *reinterpret_cast<const float4*>(
            B + (size_t)(kt + kr) * ldb + n0 + nf);
    }
}

// Convert staged fp32 -> bf16 hi/lo and commit to LDS (As[m][k], Bs[n][k]).
template <bool TRANS_A>
__device__ __forceinline__ void commit_tiles(
    int tid, const float4 (&stA)[4], const float4 (&stB)[4],
    unsigned short* As_hi, unsigned short* As_lo,
    unsigned short* Bs_hi, unsigned short* Bs_lo) {
#pragma unroll
    for (int i = 0; i < 4; ++i) {
        int f = tid + i * 256;
        float va[4] = {stA[i].x, stA[i].y, stA[i].z, stA[i].w};
        if constexpr (TRANS_A) {
            int kr = f >> 5;
            int mf = (f & 31) << 2;
#pragma unroll
            for (int j = 0; j < 4; ++j) {
                unsigned short h = f2bf(va[j]);
                As_hi[(mf + j) * LDSTRIDE + kr] = h;
                As_lo[(mf + j) * LDSTRIDE + kr] = f2bf(va[j] - bf2f(h));
            }
        } else {
            int mr = f >> 3;
            int kf = (f & 7) << 2;
#pragma unroll
            for (int j = 0; j < 4; ++j) {
                unsigned short h = f2bf(va[j]);
                As_hi[mr * LDSTRIDE + kf + j] = h;
                As_lo[mr * LDSTRIDE + kf + j] = f2bf(va[j] - bf2f(h));
            }
        }
        int kr = f >> 5;
        int nf = (f & 31) << 2;
        float vb[4] = {stB[i].x, stB[i].y, stB[i].z, stB[i].w};
#pragma unroll
        for (int j = 0; j < 4; ++j) {
            unsigned short h = f2bf(vb[j]);
            Bs_hi[(nf + j) * LDSTRIDE + kr] = h;
            Bs_lo[(nf + j) * LDSTRIDE + kr] = f2bf(vb[j] - bf2f(h));
        }
    }
}

// C[m,n] = sum_k opA(A)[m,k] * B[k,n]
//   TRANS_A=1: A stored [K,M] (A[k*lda+m]);  TRANS_A=0: A stored [M,K]
//   B always stored [K,N] row-major.
// Block tile 128x128, K-step 32. 256 threads = 8 waves in a 2(M) x 4(N) grid;
// each wave owns a 64x32 region = 4x2 WMMA 16x16 tiles.
template <bool TRANS_A>
__global__ __launch_bounds__(256) void gemm_bf16x3(
    const float* __restrict__ A, const float* __restrict__ B,
    float* __restrict__ C, int K, int lda, int ldb, int ldc,
    long long aBatch, long long bBatch, long long cBatch) {

    __shared__ unsigned short As_hi[128 * LDSTRIDE];
    __shared__ unsigned short As_lo[128 * LDSTRIDE];
    __shared__ unsigned short Bs_hi[128 * LDSTRIDE];
    __shared__ unsigned short Bs_lo[128 * LDSTRIDE];

    const int tid   = threadIdx.x;
    const int lane  = tid & 31;
    const int wave  = tid >> 5;
    const int hf    = lane >> 4;      // lane half (0: lanes 0-15, 1: 16-31)
    const int l15   = lane & 15;
    const int waveM = wave >> 2;      // 0..1
    const int waveN = wave & 3;       // 0..3

    const int m0 = blockIdx.y * 128;
    const int n0 = blockIdx.x * 128;

    A += (long long)blockIdx.z * aBatch;
    B += (long long)blockIdx.z * bBatch;
    C += (long long)blockIdx.z * cBatch;

    v8f acc[4][2];
    const v8f vzero = {0.f, 0.f, 0.f, 0.f, 0.f, 0.f, 0.f, 0.f};
#pragma unroll
    for (int mi = 0; mi < 4; ++mi)
#pragma unroll
        for (int ni = 0; ni < 2; ++ni) acc[mi][ni] = vzero;

    float4 stA[4], stB[4];
    stage_tiles<TRANS_A>(A, B, /*kt=*/0, m0, n0, lda, ldb, tid, stA, stB);

    for (int kt = 0; kt < K; kt += 32) {
        // commit staged slab (waits on its global loads here, not earlier)
        commit_tiles<TRANS_A>(tid, stA, stB, As_hi, As_lo, Bs_hi, Bs_lo);
        __syncthreads();

        // issue next slab's global loads NOW -> latency hides under WMMAs
        if (kt + 32 < K)
            stage_tiles<TRANS_A>(A, B, kt + 32, m0, n0, lda, ldb, tid, stA, stB);

        // ---- fragments from LDS (two ds_read_b128 each) ----
        v16bf a_hi[4], a_lo[4], b_hi[2], b_lo[2];
#pragma unroll
        for (int mi = 0; mi < 4; ++mi) {
            int row = waveM * 64 + mi * 16 + l15;
            a_hi[mi] = load_frag_a(&As_hi[row * LDSTRIDE], hf);
            a_lo[mi] = load_frag_a(&As_lo[row * LDSTRIDE], hf);
        }
#pragma unroll
        for (int ni = 0; ni < 2; ++ni) {
            int row = waveN * 32 + ni * 16 + l15;
            b_hi[ni] = load_frag_b(&Bs_hi[row * LDSTRIDE], hf);
            b_lo[ni] = load_frag_b(&Bs_lo[row * LDSTRIDE], hf);
        }

        // ---- 4x2 tiles x 3 products = 24 WMMAs per K-step ----
#pragma unroll
        for (int mi = 0; mi < 4; ++mi) {
#pragma unroll
            for (int ni = 0; ni < 2; ++ni) {
                acc[mi][ni] = __builtin_amdgcn_wmma_f32_16x16x32_bf16(
                    false, a_hi[mi], false, b_hi[ni], (short)0, acc[mi][ni], false, false);
                acc[mi][ni] = __builtin_amdgcn_wmma_f32_16x16x32_bf16(
                    false, a_hi[mi], false, b_lo[ni], (short)0, acc[mi][ni], false, false);
                acc[mi][ni] = __builtin_amdgcn_wmma_f32_16x16x32_bf16(
                    false, a_lo[mi], false, b_hi[ni], (short)0, acc[mi][ni], false, false);
            }
        }
        __syncthreads();  // all fragment reads done before next commit
    }

    // ---- store C: vgpr r -> M = r + 8*half, N = lane&15 ----
#pragma unroll
    for (int mi = 0; mi < 4; ++mi) {
#pragma unroll
        for (int ni = 0; ni < 2; ++ni) {
            int mbase = m0 + waveM * 64 + mi * 16 + hf * 8;
            int nn    = n0 + waveN * 32 + ni * 16 + l15;
#pragma unroll
            for (int r = 0; r < 8; ++r)
                C[(size_t)(mbase + r) * ldc + nn] = acc[mi][ni][r];
        }
    }
}

extern "C" void kernel_launch(void* const* d_in, const int* in_sizes, int n_in,
                              void* d_out, int out_size, void* d_ws, size_t ws_size,
                              hipStream_t stream) {
    const float* X = (const float*)d_in[0];  // [2, 4096, 1024] fp32
    const float* W = (const float*)d_in[1];  // [1024, 1024]    fp32
    float* out = (float*)d_out;              // [2, 4096, 1024] fp32

    constexpr int Bb = 2, Nn = 4096, Dd = 1024;
    float* G  = (float*)d_ws;                 // [2, 1024, 1024]
    float* M2 = G + (size_t)Bb * Dd * Dd;     // [2, 1024, 1024]

    dim3 blk(256);

    // Stage 1: G = X^T X   (TN: A=X as [K=4096, M=1024], B=X as [K,N])
    gemm_bf16x3<true><<<dim3(Dd / 128, Dd / 128, Bb), blk, 0, stream>>>(
        X, X, G, /*K=*/Nn, Dd, Dd, Dd,
        (long long)Nn * Dd, (long long)Nn * Dd, (long long)Dd * Dd);

    // Stage 2: M2 = W^T G  (TN: A=W as [K=1024, M=1024], shared across batch)
    gemm_bf16x3<true><<<dim3(Dd / 128, Dd / 128, Bb), blk, 0, stream>>>(
        W, G, M2, /*K=*/Dd, Dd, Dd, Dd,
        0LL, (long long)Dd * Dd, (long long)Dd * Dd);

    // Stage 3: out = X M2  (NN: A=X as [M=4096, K=1024])
    gemm_bf16x3<false><<<dim3(Dd / 128, Nn / 128, Bb), blk, 0, stream>>>(
        X, M2, out, /*K=*/Dd, Dd, Dd, Dd,
        (long long)Nn * Dd, (long long)Dd * Dd, (long long)Nn * Dd);
}
